// ExplicitSVDBlock_37812892074285
// MI455X (gfx1250) — matrix operand order repaired and checked by reading, not compiled, not run on hardware
//
#include <hip/hip_runtime.h>
#include <hip/hip_bf16.h>
#include <math.h>
#include <stdint.h>

// ---------------------------------------------------------------------------
// CDNA5 (gfx1250) wave32 WMMA transformer block, v3.1.
// - v_wmma_f32_16x16x32_bf16 for all GEMM/attention math
// - GEMM B tiles staged into LDS by the Tensor Data Mover (TENSOR_LOAD_TO_LDS,
//   6-arg clang-23 builtin form), double-buffered, synchronized with
//   s_wait_tensorcnt + barriers
// - all other fragment traffic: contiguous b128 loads
// ---------------------------------------------------------------------------

typedef __bf16 bf16;
typedef __attribute__((ext_vector_type(16))) __bf16 v16bf;
typedef __attribute__((ext_vector_type(8)))  __bf16 v8bf;
typedef __attribute__((ext_vector_type(8)))  float  v8f;
typedef unsigned int u32x4 __attribute__((ext_vector_type(4)));
typedef int          i32x8 __attribute__((ext_vector_type(8)));
typedef int          i32x4 __attribute__((ext_vector_type(4)));

#if __has_builtin(__builtin_amdgcn_tensor_load_to_lds) && __has_builtin(__builtin_amdgcn_s_wait_tensorcnt)
#define HAVE_TDM 1
#else
#define HAVE_TDM 0
#endif

#define WMMA_BF16(a, b, c) \
  __builtin_amdgcn_wmma_f32_16x16x32_bf16(false, (a), false, (b), (short)0, (c), false, false)

__device__ __forceinline__ v16bf make_v16(v8bf lo, v8bf hi) {
  return __builtin_shufflevector(lo, hi, 0, 1, 2, 3, 4, 5, 6, 7, 8, 9, 10, 11, 12, 13, 14, 15);
}

// ---------------- fragment loaders (contiguous-k order, wave32) -------------
// A (16x32 bf16), row-major [M,K]: lane row = lane%16,
// K = kb..kb+15, kb = 16*(lane/16). 32B contiguous.
__device__ __forceinline__ v16bf load_a_frag(const bf16* A, int lda, int m0, int k0, int lane) {
  int row = m0 + (lane & 15);
  int kb  = k0 + ((lane >> 4) << 4);
  const bf16* p = A + (size_t)row * lda + kb;
  v8bf lo = *(const v8bf*)p;
  v8bf hi = *(const v8bf*)(p + 8);
  return make_v16(lo, hi);
}

// B (32x16 bf16) from K-contiguous storage Bt[n][k] (i.e. B transposed):
// lane col = n0 + lane%16, K = kb..kb+15, kb = 16*(lane/16). 32B contiguous.
__device__ __forceinline__ v16bf load_bt_frag(const bf16* Bt, int ldk, int n0, int k0, int lane) {
  int col = n0 + (lane & 15);
  int kk  = k0 + ((lane >> 4) << 4);
  const bf16* p = Bt + (size_t)col * ldk + kk;
  v8bf lo = *(const v8bf*)p;
  v8bf hi = *(const v8bf*)(p + 8);
  return make_v16(lo, hi);
}

// ---------------- TDM: DMA a [64 rows x 32 k] bf16 tile of Bt into LDS ------
// D# per CDNA5 ISA: group0 {count=1, lds_addr,
// global_addr, type=2}; group1 {data_size=1(2B), tensor_dim0=K, tensor_dim1=P,
// tile_dim0=32, tile_dim1=64, tensor_dim0_stride=K}.
__device__ __forceinline__ void tdm_load_tile(const bf16* gsrc, unsigned ldsOff, int K, int P) {
#if HAVE_TDM
  unsigned long long ga = (unsigned long long)(uintptr_t)gsrc;
  u32x4 g0;
  g0[0] = 1u;                                            // count=1, user descriptor
  g0[1] = ldsOff;                                        // lds_addr (bytes)
  g0[2] = (unsigned)ga;                                  // global_addr[31:0]
  g0[3] = (unsigned)((ga >> 32) & 0x01FFFFFFu) | (2u << 30); // addr[56:32] | type=2
  i32x8 g1;
  g1[0] = 0x00010000;                                    // data_size=1 -> 2 bytes
  g1[1] = (int)(((unsigned)K & 0xFFFFu) << 16);          // tensor_dim0[15:0]
  g1[2] = (int)((((unsigned)K) >> 16) | (((unsigned)P & 0xFFFFu) << 16)); // dim0 hi | dim1 lo
  g1[3] = (int)((((unsigned)P) >> 16) | (32u << 16));    // dim1 hi | tile_dim0=32
  g1[4] = 64;                                            // tile_dim1=64, tile_dim2=0
  g1[5] = K;                                             // tensor_dim0_stride[31:0]
  g1[6] = 0;
  g1[7] = 0;
  i32x4 g2; g2[0] = 0; g2[1] = 0; g2[2] = 0; g2[3] = 0;
  i32x4 g3; g3[0] = 0; g3[1] = 0; g3[2] = 0; g3[3] = 0;
  i32x8 g4; g4[0] = 0; g4[1] = 0; g4[2] = 0; g4[3] = 0;
  g4[4] = 0; g4[5] = 0; g4[6] = 0; g4[7] = 0;
  // 6-arg form (clang-23 / therock-10.0 headers): (v4u, v8i, v4i, v4i, v8i, i32 cpol)
  __builtin_amdgcn_tensor_load_to_lds(g0, g1, g2, g3, g4, 0);
#else
  (void)gsrc; (void)ldsOff; (void)K; (void)P;
#endif
}

// ---------------- layernorm -> bf16 ----------------------------------------
__global__ void ln_bf16_kernel(const float* __restrict__ x, const float* __restrict__ g,
                               const float* __restrict__ bta, bf16* __restrict__ out, int D) {
  __shared__ float red[256];
  int row = blockIdx.x;
  const float* xr = x + (size_t)row * D;
  float s = 0.f, s2 = 0.f;
  for (int i = threadIdx.x; i < D; i += 256) { float v = xr[i]; s += v; s2 += v * v; }
  red[threadIdx.x] = s; __syncthreads();
  for (int o = 128; o > 0; o >>= 1) { if (threadIdx.x < o) red[threadIdx.x] += red[threadIdx.x + o]; __syncthreads(); }
  float mean = red[0] / D; __syncthreads();
  red[threadIdx.x] = s2; __syncthreads();
  for (int o = 128; o > 0; o >>= 1) { if (threadIdx.x < o) red[threadIdx.x] += red[threadIdx.x + o]; __syncthreads(); }
  float var  = red[0] / D - mean * mean;
  float rstd = rsqrtf(var + 1e-5f);
  for (int i = threadIdx.x; i < D; i += 256)
    out[(size_t)row * D + i] = (bf16)((xr[i] - mean) * rstd * g[i] + bta[i]);
}

// ---------------- f32 -> bf16 weight convert (optional transpose) -----------
__global__ void cvt_bf16_kernel(const float* __restrict__ src, bf16* __restrict__ dst,
                                int rows, int cols, int transpose) {
  size_t idx = (size_t)blockIdx.x * blockDim.x + threadIdx.x;
  size_t tot = (size_t)rows * cols;
  if (idx >= tot) return;
  int r = (int)(idx / cols), c = (int)(idx % cols);
  if (transpose) dst[(size_t)c * rows + r] = (bf16)src[idx];
  else           dst[idx]                  = (bf16)src[idx];
}

// ---------------- V transpose: [b,m,h,d] -> [b,h,d,m] ------------------------
__global__ void vtrans_kernel(const bf16* __restrict__ vb, bf16* __restrict__ vt,
                              int Mseq, size_t total) {
  size_t idx = (size_t)blockIdx.x * blockDim.x + threadIdx.x;
  if (idx >= total) return;
  int c = (int)(idx & 1023);
  size_t row = idx >> 10;            // b*Mseq + m
  int h = c >> 6, d = c & 63;
  int b = (int)(row / Mseq), m = (int)(row % Mseq);
  vt[(((size_t)b * 16 + h) * 64 + d) * Mseq + m] = vb[idx];
}

// ---------------- WMMA GEMM: Y[M,P] = A[M,K] @ B[K,P] (+bias)(+resid) -------
// Bt is B transposed, [P][K] row-major. Block = 8 waves = 256m x 64p.
// B tile [64p x 32k] DMA'd into LDS by TDM, double buffered.
// mode 0: store bf16 ; mode 1: store f32 ; mode 2: store f32 + resid
__global__ void gemm_bf16_wmma_kernel(const bf16* __restrict__ A, const bf16* __restrict__ Bt,
                                      const float* __restrict__ bias, void* __restrict__ Out,
                                      const float* __restrict__ resid, int Mr, int K, int P, int mode) {
  __shared__ __align__(16) bf16 tileB[2][64 * 32];
  int lane = threadIdx.x & 31;
  int wave = threadIdx.x >> 5;
  int m0 = blockIdx.y * 256 + wave * 32;
  int p0 = blockIdx.x * 64;
  v8f c0[4] = {}, c1[4] = {};
  int nsteps = K >> 5;

#if HAVE_TDM
  if (threadIdx.x == 0)
    tdm_load_tile(Bt + (size_t)p0 * K, (unsigned)(uintptr_t)&tileB[0][0], K, P);
#endif

  for (int s = 0; s < nsteps; ++s) {
    int k0  = s << 5;
    int cur = s & 1;
#if HAVE_TDM
    // overwrite-safety barrier: everyone done reading tileB[cur^1] from the previous step
    __syncthreads();
    if (s + 1 < nsteps) {
      if (threadIdx.x == 0)
        tdm_load_tile(Bt + (size_t)p0 * K + (k0 + 32), (unsigned)(uintptr_t)&tileB[cur ^ 1][0], K, P);
      __builtin_amdgcn_s_wait_tensorcnt((short)1);   // current tile done, next in flight
    } else {
      __builtin_amdgcn_s_wait_tensorcnt((short)0);
    }
    __syncthreads();                                  // tile visible to all waves
#else
    __syncthreads();
    {
      int row = threadIdx.x >> 2, q = threadIdx.x & 3; // 64 rows x 4 chunks of 8 bf16
      *(v8bf*)(&tileB[cur][row * 32 + q * 8]) =
          *(const v8bf*)(Bt + (size_t)(p0 + row) * K + k0 + q * 8);
    }
    __syncthreads();
#endif

    v16bf a0 = load_a_frag(A, K, m0,      k0, lane);
    v16bf a1 = load_a_frag(A, K, m0 + 16, k0, lane);
    const bf16* tb = &tileB[cur][0];
    int kk = (lane >> 4) << 4;
#pragma unroll
    for (int t = 0; t < 4; ++t) {
      int col = t * 16 + (lane & 15);
      v8bf lo = *(const v8bf*)(tb + col * 32 + kk);
      v8bf hi = *(const v8bf*)(tb + col * 32 + kk + 8);
      v16bf b = make_v16(lo, hi);
      c0[t] = WMMA_BF16(a0, b, c0[t]);
      c1[t] = WMMA_BF16(a1, b, c1[t]);
    }
  }

  int col = lane & 15;
  int rb  = (lane >> 4) << 3;
#pragma unroll
  for (int half = 0; half < 2; ++half) {
#pragma unroll
    for (int t = 0; t < 4; ++t) {
      v8f c = half ? c1[t] : c0[t];
      int p = p0 + t * 16 + col;
      float bv = bias ? bias[p] : 0.f;
#pragma unroll
      for (int r = 0; r < 8; ++r) {
        int m = m0 + half * 16 + rb + r;
        size_t idx = (size_t)m * P + p;
        float val = c[r] + bv;
        if (mode == 0)      ((bf16*)Out)[idx]  = (bf16)val;
        else if (mode == 1) ((float*)Out)[idx] = val;
        else                ((float*)Out)[idx] = val + resid[idx];
      }
    }
  }
}

// ---------------- RoPE on q/k (bf16 [B*M, D], head-major dims) --------------
__global__ void rope_kernel(bf16* __restrict__ q, bf16* __restrict__ k, int Mseq, size_t total) {
  size_t idx = (size_t)blockIdx.x * blockDim.x + threadIdx.x;
  if (idx >= total) return;
  int j    = (int)(idx & 31);          // freq index 0..31
  size_t g = idx >> 5;                 // (b*M+m)*H + h
  int h    = (int)(g & 15);
  size_t row = g >> 4;                 // b*M + m
  int m    = (int)(row & (size_t)(Mseq - 1));
  float invf  = __expf(-(2.f * j / 64.f) * 9.210340372f); // ln(10000)
  float ang   = (float)m * invf;
  float cs = cosf(ang), sn = sinf(ang);
  size_t base = row * 1024 + (size_t)h * 64;
  float q1 = (float)q[base + j], q2 = (float)q[base + j + 32];
  q[base + j]      = (bf16)(q1 * cs - q2 * sn);
  q[base + j + 32] = (bf16)(q2 * cs + q1 * sn);
  float k1 = (float)k[base + j], k2 = (float)k[base + j + 32];
  k[base + j]      = (bf16)(k1 * cs - k2 * sn);
  k[base + j + 32] = (bf16)(k2 * cs + k1 * sn);
}

// ---------------- flash attention: wave = 16 query rows, 32 keys/iter -------
// Q,K: [b, m, h*64+d] bf16 ; Vt: [b, h, d, m] bf16 ; O: [b, m, h*64+d] bf16
__global__ void attn_flash_kernel(const bf16* __restrict__ Q, const bf16* __restrict__ Kb,
                                  const bf16* __restrict__ Vt, bf16* __restrict__ O, int Mseq) {
  __shared__ __align__(16) bf16 plds[8 * 16 * 32]; // per-wave 16x32 P-tile buffer
  const int D = 1024;
  int lane = threadIdx.x & 31;
  int wave = threadIdx.x >> 5;
  int b = blockIdx.z, h = blockIdx.y;
  int m0 = blockIdx.x * 128 + wave * 16;
  const bf16* qp  = Q  + ((size_t)b * Mseq) * D + h * 64;
  const bf16* kp  = Kb + ((size_t)b * Mseq) * D + h * 64;
  const bf16* vtp = Vt + (((size_t)b * 16 + h) * 64) * Mseq;

  v16bf qa0 = load_a_frag(qp, D, m0, 0, lane);
  v16bf qa1 = load_a_frag(qp, D, m0, 32, lane);

  v8f acc[4] = {};
  float mrow[8], lrow[8];
#pragma unroll
  for (int r = 0; r < 8; ++r) { mrow[r] = -1e30f; lrow[r] = 0.f; }

  bf16* myl = plds + wave * (16 * 32);
  int col = lane & 15;
  int rb  = (lane >> 4) << 3;

  for (int n0 = 0; n0 < Mseq; n0 += 32) {
    v8f s0 = {}, s1 = {};
    s0 = WMMA_BF16(qa0, load_bt_frag(kp, D, n0,      0,  lane), s0);
    s0 = WMMA_BF16(qa1, load_bt_frag(kp, D, n0,      32, lane), s0);
    s1 = WMMA_BF16(qa0, load_bt_frag(kp, D, n0 + 16, 0,  lane), s1);
    s1 = WMMA_BF16(qa1, load_bt_frag(kp, D, n0 + 16, 32, lane), s1);

#pragma unroll
    for (int r = 0; r < 8; ++r) { s0[r] *= 0.125f; s1[r] *= 0.125f; }

#pragma unroll
    for (int r = 0; r < 8; ++r) {
      float mx = fmaxf(s0[r], s1[r]);
      mx = fmaxf(mx, __shfl_xor(mx, 1, 16));
      mx = fmaxf(mx, __shfl_xor(mx, 2, 16));
      mx = fmaxf(mx, __shfl_xor(mx, 4, 16));
      mx = fmaxf(mx, __shfl_xor(mx, 8, 16));
      float mnew = fmaxf(mrow[r], mx);
      float al = __expf(mrow[r] - mnew);
      float p0 = __expf(s0[r] - mnew);
      float p1 = __expf(s1[r] - mnew);
      float rs = p0 + p1;
      rs += __shfl_xor(rs, 1, 16);
      rs += __shfl_xor(rs, 2, 16);
      rs += __shfl_xor(rs, 4, 16);
      rs += __shfl_xor(rs, 8, 16);
      lrow[r] = lrow[r] * al + rs;
      mrow[r] = mnew;
      acc[0][r] *= al; acc[1][r] *= al; acc[2][r] *= al; acc[3][r] *= al;
      myl[(rb + r) * 32 + col]      = (bf16)p0;
      myl[(rb + r) * 32 + col + 16] = (bf16)p1;
    }
    asm volatile("s_wait_dscnt 0x0" ::: "memory");

    v16bf pa;
    {
      int row = lane & 15;
      int kb  = (lane >> 4) << 4;
      v8bf lo = *(const v8bf*)(myl + row * 32 + kb);
      v8bf hi = *(const v8bf*)(myl + row * 32 + kb + 8);
      pa = make_v16(lo, hi);
    }
#pragma unroll
    for (int t = 0; t < 4; ++t)
      acc[t] = WMMA_BF16(pa, load_bt_frag(vtp, Mseq, t * 16, n0, lane), acc[t]);
  }

  bf16* op = O + ((size_t)b * Mseq) * D + h * 64;
#pragma unroll
  for (int t = 0; t < 4; ++t) {
#pragma unroll
    for (int r = 0; r < 8; ++r)
      op[(size_t)(m0 + rb + r) * D + t * 16 + col] = (bf16)(acc[t][r] / lrow[r]);
  }
}

// ---------------- GEGLU (tanh gelu), 8 elements / thread --------------------
__global__ void geglu_kernel(const bf16* __restrict__ z, bf16* __restrict__ h, size_t total8) {
  size_t idx = (size_t)blockIdx.x * blockDim.x + threadIdx.x;
  if (idx >= total8) return;
  size_t r = idx / 256;
  int    c = (int)(idx % 256) * 8;
  v8bf z1 = *(const v8bf*)(z + r * 4096 + c);
  v8bf z2 = *(const v8bf*)(z + r * 4096 + 2048 + c);
  v8bf o;
#pragma unroll
  for (int i = 0; i < 8; ++i) {
    float a = (float)z1[i];
    float t = tanhf(0.7978845608f * (a + 0.044715f * a * a * a));
    o[i] = (bf16)(0.5f * a * (1.f + t) * (float)z2[i]);
  }
  *(v8bf*)(h + r * 2048 + c) = o;
}

// ---------------------------------------------------------------------------
extern "C" void kernel_launch(void* const* d_in, const int* in_sizes, int n_in,
                              void* d_out, int out_size, void* d_ws, size_t ws_size,
                              hipStream_t stream) {
  (void)in_sizes; (void)n_in; (void)out_size; (void)ws_size;
  const int Bb = 2, Mseq = 2048, D = 1024, H = 16;
  const int RA = 512, RF = 1024, F2 = 4096, FD = 2048;
  const int N = Bb * Mseq; // 4096 rows

  const float* x     = (const float*)d_in[0];
  const float* ln1g  = (const float*)d_in[1];
  const float* ln1b  = (const float*)d_in[2];
  const float* Uq    = (const float*)d_in[3];
  const float* Vq    = (const float*)d_in[4];
  const float* bq    = (const float*)d_in[5];
  const float* Uk    = (const float*)d_in[6];
  const float* Vk    = (const float*)d_in[7];
  const float* bk    = (const float*)d_in[8];
  const float* Uv    = (const float*)d_in[9];
  const float* Vv    = (const float*)d_in[10];
  const float* bv    = (const float*)d_in[11];
  const float* Wo    = (const float*)d_in[12];
  const float* bo    = (const float*)d_in[13];
  const float* ln2g  = (const float*)d_in[14];
  const float* ln2b  = (const float*)d_in[15];
  const float* U1    = (const float*)d_in[16];
  const float* V1    = (const float*)d_in[17];
  const float* b1    = (const float*)d_in[18];
  const float* U2    = (const float*)d_in[19];
  const float* V2    = (const float*)d_in[20];
  const float* b2    = (const float*)d_in[21];

  size_t off = 0;
  auto alloc = [&](size_t bytes) -> void* {
    void* p = (char*)d_ws + off;
    off += (bytes + 255) & ~(size_t)255;
    return p;
  };
  bf16* xn    = (bf16*)alloc((size_t)N * D * 2);
  bf16* wUqT  = (bf16*)alloc((size_t)D * RA * 2);
  bf16* wUkT  = (bf16*)alloc((size_t)D * RA * 2);
  bf16* wUvT  = (bf16*)alloc((size_t)D * RA * 2);
  bf16* wVqT  = (bf16*)alloc((size_t)RA * D * 2);
  bf16* wVkT  = (bf16*)alloc((size_t)RA * D * 2);
  bf16* wVvT  = (bf16*)alloc((size_t)RA * D * 2);
  bf16* wWo   = (bf16*)alloc((size_t)D * D * 2);   // B^T for @Wo^T is Wo itself
  bf16* wU1T  = (bf16*)alloc((size_t)D * RF * 2);
  bf16* wV1T  = (bf16*)alloc((size_t)RF * F2 * 2);
  bf16* wU2T  = (bf16*)alloc((size_t)FD * RF * 2);
  bf16* wV2T  = (bf16*)alloc((size_t)RF * D * 2);
  bf16* tq    = (bf16*)alloc((size_t)N * RA * 2);
  bf16* tk    = (bf16*)alloc((size_t)N * RA * 2);
  bf16* tv    = (bf16*)alloc((size_t)N * RA * 2);
  bf16* qb    = (bf16*)alloc((size_t)N * D * 2);
  bf16* kb    = (bf16*)alloc((size_t)N * D * 2);
  bf16* vb    = (bf16*)alloc((size_t)N * D * 2);
  bf16* vt    = (bf16*)alloc((size_t)N * D * 2);
  bf16* ao    = (bf16*)alloc((size_t)N * D * 2);
  float* x1   = (float*)alloc((size_t)N * D * 4);
  bf16* xn2   = (bf16*)alloc((size_t)N * D * 2);
  bf16* t1    = (bf16*)alloc((size_t)N * RF * 2);
  bf16* z     = (bf16*)alloc((size_t)N * F2 * 2);
  bf16* hb    = (bf16*)alloc((size_t)N * FD * 2);
  bf16* t2    = (bf16*)alloc((size_t)N * RF * 2);

  auto cvt = [&](const float* s, bf16* d, int r, int c, int tr) {
    size_t tot = (size_t)r * c;
    cvt_bf16_kernel<<<(unsigned)((tot + 255) / 256), 256, 0, stream>>>(s, d, r, c, tr);
  };
  auto gemm = [&](const bf16* A, const bf16* Bt, const float* bias, void* O,
                  const float* res, int Mr, int K, int P, int mode) {
    dim3 g(P / 64, Mr / 256);
    gemm_bf16_wmma_kernel<<<g, 256, 0, stream>>>(A, Bt, bias, O, res, Mr, K, P, mode);
  };

  // weight conversion: all B operands stored K-contiguous (transposed),
  // except Wo whose transpose-of-transpose is Wo itself.
  cvt(Uq, wUqT, D, RA, 1);  cvt(Uk, wUkT, D, RA, 1);  cvt(Uv, wUvT, D, RA, 1);
  cvt(Vq, wVqT, RA, D, 1);  cvt(Vk, wVkT, RA, D, 1);  cvt(Vv, wVvT, RA, D, 1);
  cvt(Wo, wWo, D, D, 0);
  cvt(U1, wU1T, D, RF, 1);  cvt(V1, wV1T, RF, F2, 1);
  cvt(U2, wU2T, FD, RF, 1); cvt(V2, wV2T, RF, D, 1);

  // LN1
  ln_bf16_kernel<<<N, 256, 0, stream>>>(x, ln1g, ln1b, xn, D);

  // low-rank QKV
  gemm(xn, wUqT, nullptr, tq, nullptr, N, D, RA, 0);
  gemm(tq, wVqT, bq,      qb, nullptr, N, RA, D, 0);
  gemm(xn, wUkT, nullptr, tk, nullptr, N, D, RA, 0);
  gemm(tk, wVkT, bk,      kb, nullptr, N, RA, D, 0);
  gemm(xn, wUvT, nullptr, tv, nullptr, N, D, RA, 0);
  gemm(tv, wVvT, bv,      vb, nullptr, N, RA, D, 0);

  // RoPE on q,k
  {
    size_t tot = (size_t)N * H * 32;
    rope_kernel<<<(unsigned)((tot + 255) / 256), 256, 0, stream>>>(qb, kb, Mseq, tot);
  }
  // V -> V^T ([b,h,d,m]) so P@V fragments are contiguous
  {
    size_t tot = (size_t)N * D;
    vtrans_kernel<<<(unsigned)((tot + 255) / 256), 256, 0, stream>>>(vb, vt, Mseq, tot);
  }

  // flash attention
  {
    dim3 g(Mseq / 128, H, Bb);
    attn_flash_kernel<<<g, 256, 0, stream>>>(qb, kb, vt, ao, Mseq);
  }

  // x1 = x + ao @ Wo^T + bo
  gemm(ao, wWo, bo, x1, x, N, D, D, 2);

  // LN2 + GEGLU FFN
  ln_bf16_kernel<<<N, 256, 0, stream>>>(x1, ln2g, ln2b, xn2, D);
  gemm(xn2, wU1T, nullptr, t1, nullptr, N, D, RF, 0);
  gemm(t1, wV1T, b1, z, nullptr, N, RF, F2, 0);
  {
    size_t tot8 = (size_t)N * FD / 8;
    geglu_kernel<<<(unsigned)((tot8 + 255) / 256), 256, 0, stream>>>(z, hb, tot8);
  }
  gemm(hb, wU2T, nullptr, t2, nullptr, N, FD, RF, 0);
  // out = x1 + t2 @ V2 + b2
  gemm(t2, wV2T, b2, d_out, x1, N, RF, D, 2);
}
